// HGTModel_85993835200826
// MI455X (gfx1250) — compile-verified, matrix operand order checked by reading, not run on hardware
//
#include <hip/hip_runtime.h>
#include <hip/hip_bf16.h>
#include <math.h>

#define NODES   50000
#define E_EDGES 200000
#define HEADS   8
#define DHEAD   64
#define CIN     256
#define CH      512

typedef __attribute__((ext_vector_type(16))) __bf16 v16bf;
typedef __attribute__((ext_vector_type(8)))  float  v8f;

// ---------------- helpers ----------------

__device__ __forceinline__ float gelu_f(float x) {
  // exact gelu: 0.5*x*(1+erf(x/sqrt(2)))
  return 0.5f * x * (1.0f + erff(x * 0.70710678118654752440f));
}

// order-preserving float <-> uint for atomicMax on floats (handles negatives)
__device__ __forceinline__ unsigned flip_f32(float f) {
  unsigned u = __float_as_uint(f);
  return (u & 0x80000000u) ? ~u : (u | 0x80000000u);
}
__device__ __forceinline__ float unflip_f32(unsigned u) {
  u = (u & 0x80000000u) ? (u & 0x7fffffffu) : ~u;
  return __uint_as_float(u);
}

__device__ __forceinline__ v16bf pack16_bf16(float4 a0, float4 a1, float4 a2, float4 a3) {
  v16bf v;
  v[0]  = (__bf16)a0.x; v[1]  = (__bf16)a0.y; v[2]  = (__bf16)a0.z; v[3]  = (__bf16)a0.w;
  v[4]  = (__bf16)a1.x; v[5]  = (__bf16)a1.y; v[6]  = (__bf16)a1.z; v[7]  = (__bf16)a1.w;
  v[8]  = (__bf16)a2.x; v[9]  = (__bf16)a2.y; v[10] = (__bf16)a2.z; v[11] = (__bf16)a2.w;
  v[12] = (__bf16)a3.x; v[13] = (__bf16)a3.y; v[14] = (__bf16)a3.z; v[15] = (__bf16)a3.w;
  return v;
}

// ---------------- fills ----------------

__global__ void fill_f32_k(float* __restrict__ p, float v, int n) {
  int i = blockIdx.x * blockDim.x + threadIdx.x;
  if (i < n) p[i] = v;
}
__global__ void fill_u32_k(unsigned* __restrict__ p, unsigned v, int n) {
  int i = blockIdx.x * blockDim.x + threadIdx.x;
  if (i < n) p[i] = v;
}

// ---------------- dense GEMM: Out[M,N] = epi(pro(A)[M,K] @ Wt[K,N] + bias) ----------------
// One 16x64 output strip per wave32 (4 WMMA accumulators; A fragment loaded &
// converted once per K-step, reused 4x). 8 waves/block cover N = 512.
// grid = (M/16), block = 256.
// PRO: 0 = identity, 1 = gelu(A).  EPI: 0 = bias, 1 = bias+relu, 2 = bias+gated-skip.
template <int PRO, int EPI>
__global__ void gemm_wmma_k(const float* __restrict__ A, const float* __restrict__ Wt,
                            const float* __restrict__ bias,
                            const float* __restrict__ skipX, const float* __restrict__ gate,
                            float* __restrict__ Out, int K, int N) {
  const int wave  = threadIdx.x >> 5;    // 0..7 -> 64-column strip
  const int lane  = threadIdx.x & 31;
  const int tileM = blockIdx.x;
  const int row   = lane & 15;
  const int khalf = lane >> 4;           // 0: K 0..7 / 16..23, 1: K 8..15 / 24..31
  const int rowg  = tileM * 16 + row;    // A row for this lane
  const int col0  = wave * 64 + row;     // first of 4 B/C columns for this lane
  v8f c0 = {0.f, 0.f, 0.f, 0.f, 0.f, 0.f, 0.f, 0.f};
  v8f c1 = c0, c2 = c0, c3 = c0;

  for (int kk = 0; kk < K; kk += 32) {
    const float* ab = A + (size_t)rowg * K + kk + khalf * 8;
    float4 a0 = *(const float4*)(ab);
    float4 a1 = *(const float4*)(ab + 4);
    float4 a2 = *(const float4*)(ab + 16);
    float4 a3 = *(const float4*)(ab + 20);
    if (PRO == 1) {
      a0.x = gelu_f(a0.x); a0.y = gelu_f(a0.y); a0.z = gelu_f(a0.z); a0.w = gelu_f(a0.w);
      a1.x = gelu_f(a1.x); a1.y = gelu_f(a1.y); a1.z = gelu_f(a1.z); a1.w = gelu_f(a1.w);
      a2.x = gelu_f(a2.x); a2.y = gelu_f(a2.y); a2.z = gelu_f(a2.z); a2.w = gelu_f(a2.w);
      a3.x = gelu_f(a3.x); a3.y = gelu_f(a3.y); a3.z = gelu_f(a3.z); a3.w = gelu_f(a3.w);
    }
    const v16bf av = pack16_bf16(a0, a1, a2, a3);
#pragma unroll
    for (int t = 0; t < 4; ++t) {
      v16bf bv;
#pragma unroll
      for (int j = 0; j < 16; ++j) {
        int kb = kk + (j < 8 ? j : j + 8) + khalf * 8;
        bv[j] = (__bf16)Wt[(size_t)kb * N + col0 + t * 16];
      }
      v8f& c = (t == 0) ? c0 : (t == 1) ? c1 : (t == 2) ? c2 : c3;
      c = __builtin_amdgcn_wmma_f32_16x16x32_bf16(false, av, false, bv, (short)0, c,
                                                  false, false);
    }
  }

  float s = 0.f;
  if (EPI == 2) s = 1.f / (1.f + expf(-gate[0]));
#pragma unroll
  for (int t = 0; t < 4; ++t) {
    const v8f& c = (t == 0) ? c0 : (t == 1) ? c1 : (t == 2) ? c2 : c3;
    const int coln = col0 + t * 16;
    const float bcol = bias[coln];
#pragma unroll
    for (int j = 0; j < 8; ++j) {
      int r = tileM * 16 + j + khalf * 8;
      float v = c[j] + bcol;
      if (EPI == 1) v = fmaxf(v, 0.f);
      if (EPI == 2) v = s * v + (1.f - s) * skipX[(size_t)r * N + coln];
      Out[(size_t)r * N + coln] = v;
    }
  }
}

// ---------------- block-diagonal per-head transform ----------------
// Out[n, h*64+e] = sum_d A[n, h*64+d] * R[h, d, e];  A,Out: [M,512], R: [8,64,64]
// wave = head; each wave produces the head's full 16x64 strip (4 accumulators,
// K = 64 in 2 steps, fully unrolled -> 8 static WMMAs). grid = (M/16), block = 256.
__global__ void gemm_rel_k(const float* __restrict__ A, const float* __restrict__ R,
                           float* __restrict__ Out) {
  const int head  = threadIdx.x >> 5;             // 0..7
  const int lane  = threadIdx.x & 31;
  const int tileM = blockIdx.x;
  const int row   = lane & 15;
  const int khalf = lane >> 4;
  const int rowg  = tileM * 16 + row;
  const int colL  = lane & 15;                    // + t*16 -> local col in head
  v8f c0 = {0.f, 0.f, 0.f, 0.f, 0.f, 0.f, 0.f, 0.f};
  v8f c1 = c0, c2 = c0, c3 = c0;

#pragma unroll
  for (int kk = 0; kk < DHEAD; kk += 32) {
    const float* ab = A + (size_t)rowg * CH + head * DHEAD + kk + khalf * 8;
    float4 a0 = *(const float4*)(ab);
    float4 a1 = *(const float4*)(ab + 4);
    float4 a2 = *(const float4*)(ab + 16);
    float4 a3 = *(const float4*)(ab + 20);
    const v16bf av = pack16_bf16(a0, a1, a2, a3);
#pragma unroll
    for (int t = 0; t < 4; ++t) {
      v16bf bv;
#pragma unroll
      for (int j = 0; j < 16; ++j) {
        int kb = kk + (j < 8 ? j : j + 8) + khalf * 8;
        bv[j] = (__bf16)R[head * (DHEAD * DHEAD) + kb * DHEAD + t * 16 + colL];
      }
      v8f& c = (t == 0) ? c0 : (t == 1) ? c1 : (t == 2) ? c2 : c3;
      c = __builtin_amdgcn_wmma_f32_16x16x32_bf16(false, av, false, bv, (short)0, c,
                                                  false, false);
    }
  }
#pragma unroll
  for (int t = 0; t < 4; ++t) {
    const v8f& c = (t == 0) ? c0 : (t == 1) ? c1 : (t == 2) ? c2 : c3;
#pragma unroll
    for (int j = 0; j < 8; ++j) {
      int r = tileM * 16 + j + khalf * 8;
      Out[(size_t)r * CH + head * DHEAD + t * 16 + colL] = c[j];
    }
  }
}

// ---------------- edge phase ----------------
// alpha[e,h] = dot(q[dst,h,:], krel[src,h,:]) * prior[h] / sqrt(64); atomic segment max
__global__ void edge_alpha_max_k(const float* __restrict__ q, const float* __restrict__ krel,
                                 const int* __restrict__ ei, const float* __restrict__ prior,
                                 float* __restrict__ alpha, unsigned* __restrict__ smax,
                                 int nE) {
  int idx = blockIdx.x * blockDim.x + threadIdx.x;
  if (idx >= nE * HEADS) return;
  int e = idx >> 3, h = idx & 7;
  int src = ei[e], dst = ei[nE + e];
  const float4* qp = (const float4*)(q + (size_t)dst * CH + h * DHEAD);
  const float4* kp = (const float4*)(krel + (size_t)src * CH + h * DHEAD);
  float acc = 0.f;
#pragma unroll
  for (int i = 0; i < 16; ++i) {
    float4 a = qp[i], b = kp[i];
    acc += a.x * b.x + a.y * b.y + a.z * b.z + a.w * b.w;
  }
  float al = acc * prior[h] * 0.125f;   // 1/sqrt(64)
  alpha[idx] = al;
  atomicMax(smax + (size_t)dst * HEADS + h, flip_f32(al));
}

// e = exp(alpha - max); atomic segment sum
__global__ void edge_expsum_k(float* __restrict__ alpha, const unsigned* __restrict__ smax,
                              float* __restrict__ ssum, const int* __restrict__ ei, int nE) {
  int idx = blockIdx.x * blockDim.x + threadIdx.x;
  if (idx >= nE * HEADS) return;
  int e = idx >> 3, h = idx & 7;
  int dst = ei[nE + e];
  float m = unflip_f32(smax[(size_t)dst * HEADS + h]);
  if (!(m >= -3.0e38f && m <= 3.0e38f)) m = 0.f;   // mirrors where(isfinite(m), m, 0)
  float ex = expf(alpha[idx] - m);
  alpha[idx] = ex;
  atomicAdd(ssum + (size_t)dst * HEADS + h, ex);
}

// agg[dst, c] += (alpha/(sum+1e-16)) * vrel[src, c]
__global__ void edge_msg_k(const float* __restrict__ alpha, const float* __restrict__ ssum,
                           const float* __restrict__ vrel, const int* __restrict__ ei,
                           float* __restrict__ agg, int nE) {
  int idx = blockIdx.x * blockDim.x + threadIdx.x;
  if (idx >= nE * CH) return;
  int e = idx >> 9, c = idx & (CH - 1);
  int h = c >> 6;
  int src = ei[e], dst = ei[nE + e];
  float w = alpha[(size_t)e * HEADS + h] / (ssum[(size_t)dst * HEADS + h] + 1e-16f);
  atomicAdd(agg + (size_t)dst * CH + c, w * vrel[(size_t)src * CH + c]);
}

// ---------------- final linear: out[M,3] = h @ lin_w + lin_b ----------------
__global__ void final_lin_k(const float* __restrict__ h, const float* __restrict__ lw,
                            const float* __restrict__ lb, float* __restrict__ out, int M) {
  int n = blockIdx.x * blockDim.x + threadIdx.x;
  if (n >= M) return;
  float a0 = lb[0], a1 = lb[1], a2 = lb[2];
  const float* hp = h + (size_t)n * CH;
  for (int d = 0; d < CH; ++d) {
    float x = hp[d];
    a0 += x * lw[d * 3 + 0];
    a1 += x * lw[d * 3 + 1];
    a2 += x * lw[d * 3 + 2];
  }
  out[n * 3 + 0] = a0;
  out[n * 3 + 1] = a1;
  out[n * 3 + 2] = a2;
}

// ---------------- host launch ----------------

extern "C" void kernel_launch(void* const* d_in, const int* in_sizes, int n_in,
                              void* d_out, int out_size, void* d_ws, size_t ws_size,
                              hipStream_t stream) {
  (void)in_sizes; (void)n_in; (void)out_size; (void)ws_size;
  auto P = [&](int i) { return (const float*)d_in[i]; };
  const float* x_user = P(0);
  const float* x_item = P(1);
  const int* e_u2i = (const int*)d_in[2];
  const int* e_i2u = (const int*)d_in[3];
  // params (insertion order): c1 base 4, c2 base 28, lin at 52/53

  const size_t NC = (size_t)NODES * CH;
  float* ws     = (float*)d_ws;
  float* k_u    = ws + 0 * NC;
  float* q_u    = ws + 1 * NC;
  float* v_u    = ws + 2 * NC;
  float* k_i    = ws + 3 * NC;
  float* q_i    = ws + 4 * NC;
  float* v_i    = ws + 5 * NC;
  float* krel_u = ws + 6 * NC;
  float* vrel_u = ws + 7 * NC;
  float* krel_i = ws + 8 * NC;
  float* vrel_i = ws + 9 * NC;
  float* agg_u  = ws + 10 * NC;
  float* agg_i  = ws + 11 * NC;
  float* h_u    = ws + 12 * NC;
  float* h_i    = ws + 13 * NC;
  float* h2_u   = agg_i;  // agg_item unused in layer 2 -> reuse for layer-2 user output
  float*    alpha = ws + 14 * NC;                                  // E*H
  unsigned* smax  = (unsigned*)(alpha + (size_t)E_EDGES * HEADS);  // N*H
  float*    ssum  = (float*)(smax + (size_t)NODES * HEADS);        // N*H

  const dim3 blk(256);
  const dim3 grd(NODES / 16);   // 3125 blocks; 8 waves x 64 cols = full N = 512
  const int nNC = (int)NC;
  const int nNH = NODES * HEADS;
  const int nEH = E_EDGES * HEADS;
  const int nEC = E_EDGES * CH;

  auto relation_pass = [&](const int* ei, const float* qdst, const float* kr,
                           const float* vr, const float* prior, float* agg) {
    fill_u32_k<<<(nNH + 255) / 256, 256, 0, stream>>>(smax, 0x007FFFFFu, nNH); // flip(-inf)
    fill_f32_k<<<(nNH + 255) / 256, 256, 0, stream>>>(ssum, 0.f, nNH);
    edge_alpha_max_k<<<(nEH + 255) / 256, 256, 0, stream>>>(qdst, kr, ei, prior, alpha,
                                                            smax, E_EDGES);
    edge_expsum_k<<<(nEH + 255) / 256, 256, 0, stream>>>(alpha, smax, ssum, ei, E_EDGES);
    edge_msg_k<<<(nEC + 255) / 256, 256, 0, stream>>>(alpha, ssum, vr, ei, agg, E_EDGES);
  };

  // -------- Layer 1 (K = 256) --------
  gemm_wmma_k<0, 0><<<grd, blk, 0, stream>>>(x_user, P(4),  P(5),  nullptr, nullptr, k_u, CIN, CH);
  gemm_wmma_k<0, 0><<<grd, blk, 0, stream>>>(x_user, P(6),  P(7),  nullptr, nullptr, q_u, CIN, CH);
  gemm_wmma_k<0, 0><<<grd, blk, 0, stream>>>(x_user, P(8),  P(9),  nullptr, nullptr, v_u, CIN, CH);
  gemm_wmma_k<0, 0><<<grd, blk, 0, stream>>>(x_item, P(13), P(14), nullptr, nullptr, k_i, CIN, CH);
  gemm_wmma_k<0, 0><<<grd, blk, 0, stream>>>(x_item, P(15), P(16), nullptr, nullptr, q_i, CIN, CH);
  gemm_wmma_k<0, 0><<<grd, blk, 0, stream>>>(x_item, P(17), P(18), nullptr, nullptr, v_i, CIN, CH);
  gemm_rel_k<<<grd, blk, 0, stream>>>(k_u, P(22), krel_u);   // arel_u2i
  gemm_rel_k<<<grd, blk, 0, stream>>>(v_u, P(23), vrel_u);   // mrel_u2i
  gemm_rel_k<<<grd, blk, 0, stream>>>(k_i, P(25), krel_i);   // arel_i2u
  gemm_rel_k<<<grd, blk, 0, stream>>>(v_i, P(26), vrel_i);   // mrel_i2u
  fill_f32_k<<<(nNC + 255) / 256, 256, 0, stream>>>(agg_i, 0.f, nNC);
  fill_f32_k<<<(nNC + 255) / 256, 256, 0, stream>>>(agg_u, 0.f, nNC);
  relation_pass(e_u2i, q_i, krel_u, vrel_u, P(24), agg_i);   // u2i: dst = item
  relation_pass(e_i2u, q_u, krel_i, vrel_i, P(27), agg_u);   // i2u: dst = user
  // out proj: gelu prologue, relu epilogue (h = relu(conv1 out))
  gemm_wmma_k<1, 1><<<grd, blk, 0, stream>>>(agg_u, P(10), P(11), nullptr, nullptr, h_u, CH, CH);
  gemm_wmma_k<1, 1><<<grd, blk, 0, stream>>>(agg_i, P(19), P(20), nullptr, nullptr, h_i, CH, CH);

  // -------- Layer 2 (K = 512): only i2u path feeds the final user output --------
  gemm_wmma_k<0, 0><<<grd, blk, 0, stream>>>(h_u, P(30), P(31), nullptr, nullptr, q_u, CH, CH);
  gemm_wmma_k<0, 0><<<grd, blk, 0, stream>>>(h_i, P(37), P(38), nullptr, nullptr, k_i, CH, CH);
  gemm_wmma_k<0, 0><<<grd, blk, 0, stream>>>(h_i, P(41), P(42), nullptr, nullptr, v_i, CH, CH);
  gemm_rel_k<<<grd, blk, 0, stream>>>(k_i, P(49), krel_i);   // c2.arel_i2u
  gemm_rel_k<<<grd, blk, 0, stream>>>(v_i, P(50), vrel_i);   // c2.mrel_i2u
  fill_f32_k<<<(nNC + 255) / 256, 256, 0, stream>>>(agg_u, 0.f, nNC);
  relation_pass(e_i2u, q_u, krel_i, vrel_i, P(51), agg_u);
  // out proj: gelu prologue, gated-skip epilogue (skip x = h_u, gate = c2.skip_user)
  gemm_wmma_k<1, 2><<<grd, blk, 0, stream>>>(agg_u, P(34), P(35), h_u, P(36), h2_u, CH, CH);

  // -------- final classifier --------
  final_lin_k<<<(NODES + 255) / 256, 256, 0, stream>>>(h2_u, P(52), P(53), (float*)d_out,
                                                       NODES);
}
